// CompressedSparseLayerELSA_22737556865333
// MI455X (gfx1250) — matrix operand order repaired and checked by reading, not compiled, not run on hardware
//
#include <hip/hip_runtime.h>

// ---------------------------------------------------------------------------
// Problem constants (from reference):
//   x  : [1024, 50000] f32     A : [50000, 256] f32
//   out: [1024, 50000] f32 = relu(x @ Am @ Am.T - x),
//   Am = row-normalize(top-64-by-|v| mask of A)
// ---------------------------------------------------------------------------
#define BATCH   1024
#define NITEMS  50000
#define NDIMS   256
#define TOPK    64
#define KSPLIT  8

typedef __attribute__((ext_vector_type(16))) __bf16 v16bf;
typedef __attribute__((ext_vector_type(8)))  __bf16 v8bf;
typedef __attribute__((ext_vector_type(4)))  __bf16 v4bf;
typedef __attribute__((ext_vector_type(8)))  float  v8f;
typedef __attribute__((ext_vector_type(4)))  float  v4f;

__device__ __forceinline__ __bf16 f2bf(float f) { return (__bf16)f; }

__device__ __forceinline__ int wave_sum_i(int v) {
  #pragma unroll
  for (int off = 16; off; off >>= 1) v += __shfl_xor(v, off, 32);
  return v;
}
__device__ __forceinline__ float wave_sum_f(float v) {
  #pragma unroll
  for (int off = 16; off; off >>= 1) v += __shfl_xor(v, off, 32);
  return v;
}

__device__ __forceinline__ v16bf cat8(v8bf a0, v8bf a1) {
  v16bf a;
  #pragma unroll
  for (int j = 0; j < 8; ++j) { a[j] = a0[j]; a[8 + j] = a1[j]; }
  return a;
}

// ---------------------------------------------------------------------------
// Kernel 1: per-row top-64 |value| mask + L2 normalize, emit bf16 Am row-major.
// One wave per row (256 cols -> 8 values per lane). Threshold found via
// wave-uniform binary search over positive-float bit patterns (monotone).
// ---------------------------------------------------------------------------
__global__ void k_mask_norm(const float* __restrict__ A, __bf16* __restrict__ Am) {
  const int lane = threadIdx.x & 31;
  const int row  = blockIdx.x * 8 + (threadIdx.x >> 5);   // 6250 blocks * 8 waves
  const float* p = A + (size_t)row * NDIMS + lane * 8;

  float v[8];
  unsigned av[8];
  v4f lo4 = *(const v4f*)p;
  v4f hi4 = *(const v4f*)(p + 4);
  #pragma unroll
  for (int j = 0; j < 4; ++j) { v[j] = lo4[j]; v[4 + j] = hi4[j]; }
  #pragma unroll
  for (int j = 0; j < 8; ++j) {
    union { float f; unsigned u; } cv; cv.f = v[j];
    av[j] = cv.u & 0x7FFFFFFFu;
  }

  // binary search: largest T with count(|a| >= T) >= TOPK  -> T = 64th largest
  unsigned tlo = 0u, thi = 0x7F800000u;
  while (tlo < thi) {
    unsigned mid = tlo + ((thi - tlo + 1u) >> 1);
    int c = 0;
    #pragma unroll
    for (int j = 0; j < 8; ++j) c += (av[j] >= mid) ? 1 : 0;
    c = wave_sum_i(c);
    if (c >= TOPK) tlo = mid; else thi = mid - 1u;
  }
  const unsigned thr = tlo;

  float ss = 0.f;
  #pragma unroll
  for (int j = 0; j < 8; ++j) {
    v[j] = (av[j] >= thr) ? v[j] : 0.0f;
    ss += v[j] * v[j];
  }
  ss = wave_sum_f(ss);
  const float scale = 1.0f / fmaxf(sqrtf(ss), 1e-12f);

  __bf16* q = Am + (size_t)row * NDIMS + lane * 8;
  #pragma unroll
  for (int j = 0; j < 8; ++j) q[j] = f2bf(v[j] * scale);
}

// ---------------------------------------------------------------------------
// Kernel 2: LDS-tiled transpose Am[50000,256] -> AmT[256,50000] (bf16), so
// GEMM1 B-fragments are K-contiguous. 16 rows x 256 cols per block, padded
// LDS to kill bank conflicts on the transposed read.
// ---------------------------------------------------------------------------
__global__ void k_transpose(const __bf16* __restrict__ Am, __bf16* __restrict__ AmT) {
  __shared__ __bf16 tile[16][NDIMS + 8];
  const int i0 = blockIdx.x * 16;                 // 3125 blocks
  const int t  = threadIdx.x;                     // 256 threads
  #pragma unroll
  for (int k = 0; k < 16; ++k)
    tile[k][t] = Am[(size_t)(i0 + k) * NDIMS + t];
  __syncthreads();
  const int j  = t & 15;        // item offset within tile (fast across lanes)
  const int d0 = t >> 4;
  #pragma unroll
  for (int d = d0; d < NDIMS; d += 16)
    AmT[(size_t)d * NITEMS + i0 + j] = tile[j][d];
}

// ---------------------------------------------------------------------------
// Kernel 3: GEMM1  xA = x @ Am   (M=1024, N=256, K=50000), bf16 WMMA.
// grid = (64 M-tiles, KSPLIT). Block = 256 thr = 8 waves; wave w owns cols
// [w*32, w*32+32). The shared 16x64 x-chunk is staged through LDS as bf16,
// double-buffered, with the *issue* (global load -> VGPRs) placed before the
// compute body and the *commit* (cvt + ds_store) after it, so the stage load
// latency hides behind the 4 WMMAs + B loads of the current iteration.
// Partials per kc (no atomics -> deterministic) written to ws.
// ---------------------------------------------------------------------------
#define G1_ROWPAD 72   // 64 + 8 bf16 pad: bank-conflict-free column reads

__device__ __forceinline__ v4f issue_x_tile(const float* __restrict__ xstripe,
                                            int k, int t) {
  const int r = t >> 4;
  const int c = (t & 15) * 4;
  const float* src = xstripe + (size_t)r * NITEMS + k + c;
  __builtin_prefetch(src + KSPLIT * 64, 0, 1);       // global_prefetch_b8
  return *(const v4f*)src;
}

__device__ __forceinline__ void commit_x_tile(v4f f, __bf16* __restrict__ buf,
                                              int t) {
  const int r = t >> 4;
  const int c = (t & 15) * 4;
  v4bf b;
  #pragma unroll
  for (int j = 0; j < 4; ++j) b[j] = f2bf(f[j]);
  *(v4bf*)(buf + r * G1_ROWPAD + c) = b;
}

__global__ void k_gemm1(const float* __restrict__ x, const __bf16* __restrict__ AmT,
                        float* __restrict__ part) {
  __shared__ __bf16 abuf[2][16 * G1_ROWPAD];
  const int t    = threadIdx.x;
  const int lane = t & 31;
  const int wv   = t >> 5;
  const int hi   = lane >> 4;           // 0: low half lanes, 1: high half
  const int l15  = lane & 15;
  const int m0   = blockIdx.x * 16;
  const int kc   = blockIdx.y;
  const int n0   = wv * 32;

  const float*  xstripe = x   + (size_t)m0 * NITEMS;
  const float*  xrow    = x   + (size_t)(m0 + l15) * NITEMS;   // for the tail
  const __bf16* bcol0   = AmT + (size_t)(n0 + l15)      * NITEMS;
  const __bf16* bcol1   = AmT + (size_t)(n0 + 16 + l15) * NITEMS;

  v8f c0 = {}; v8f c1 = {};

  int cb = 0;
  commit_x_tile(issue_x_tile(xstripe, kc * 64, t), abuf[0], t);
  __syncthreads();

  for (int c = kc; c * 64 + 64 <= NITEMS; c += KSPLIT) {
    const int k = c * 64;
    const bool more = (c + KSPLIT) * 64 + 64 <= NITEMS;

    v4f nf = {};
    if (more) nf = issue_x_tile(xstripe, (c + KSPLIT) * 64, t);  // load in flight

    #pragma unroll
    for (int s = 0; s < 2; ++s) {                       // two K=32 steps
      const __bf16* ap = abuf[cb] + l15 * G1_ROWPAD + s * 32 + hi * 16;
      v8bf a0 = *(const v8bf*)ap;
      v8bf a1 = *(const v8bf*)(ap + 8);
      v16bf a = cat8(a0, a1);
      v16bf b0 = *(const v16bf*)(bcol0 + k + s * 32 + hi * 16);
      v16bf b1 = *(const v16bf*)(bcol1 + k + s * 32 + hi * 16);
      c0 = __builtin_amdgcn_wmma_f32_16x16x32_bf16(false, a, false, b0, (short)0, c0, false, false);
      c1 = __builtin_amdgcn_wmma_f32_16x16x32_bf16(false, a, false, b1, (short)0, c1, false, false);
    }

    if (more) commit_x_tile(nf, abuf[cb ^ 1], t);       // cvt + ds_store late

    __syncthreads();
    cb ^= 1;
  }

  // K tail: 50000 % 64 == 16 elements, owned by exactly one kc; direct global.
  const int ctail = NITEMS / 64;
  const int ktail = ctail * 64;
  if (ktail < NITEMS && ((ctail % KSPLIT) == kc)) {
    v16bf a, b0, b1;
    #pragma unroll
    for (int j = 0; j < 8; ++j) {
      int ka = ktail + hi * 16 + j;
      int kb = ka + 8;
      a[j]     = (ka < NITEMS) ? f2bf(xrow[ka]) : f2bf(0.0f);
      a[8 + j] = (kb < NITEMS) ? f2bf(xrow[kb]) : f2bf(0.0f);
    }
    #pragma unroll
    for (int j = 0; j < 16; ++j) {
      int kk = ktail + hi * 16 + j;
      b0[j] = (kk < NITEMS) ? bcol0[kk] : f2bf(0.0f);
      b1[j] = (kk < NITEMS) ? bcol1[kk] : f2bf(0.0f);
    }
    c0 = __builtin_amdgcn_wmma_f32_16x16x32_bf16(false, a, false, b0, (short)0, c0, false, false);
    c1 = __builtin_amdgcn_wmma_f32_16x16x32_bf16(false, a, false, b1, (short)0, c1, false, false);
  }

  float* pp = part + (size_t)kc * (BATCH * NDIMS);
  #pragma unroll
  for (int r = 0; r < 8; ++r) {
    const int row = m0 + r + 8 * hi;
    pp[(size_t)row * NDIMS + n0 + l15]      = c0[r];
    pp[(size_t)row * NDIMS + n0 + 16 + l15] = c1[r];
  }
}

// Deterministic split-K reduction: xA = sum_kc part[kc] (fixed order).
__global__ void k_reduce(const float* __restrict__ part, float* __restrict__ xA) {
  const int idx = blockIdx.x * blockDim.x + threadIdx.x;   // 262144 elems
  float s = 0.f;
  #pragma unroll
  for (int kc = 0; kc < KSPLIT; ++kc) s += part[(size_t)kc * (BATCH * NDIMS) + idx];
  xA[idx] = s;
}

// ---------------------------------------------------------------------------
// Kernel 4: GEMM2  out = relu(xA @ Am.T - x)  (M=1024, N=50000, K=256).
// The shared 16x256 xA stripe is tiny (8KB bf16): staged once into LDS by all
// 256 threads, single barrier, then each wave runs 8 unrolled WMMA steps with
// A-frags from LDS and B-frags (rows of row-major bf16 Am) from global.
// Fused subtract + relu epilogue.
// ---------------------------------------------------------------------------
__global__ void k_gemm2(const float* __restrict__ xA, const __bf16* __restrict__ Am,
                        const float* __restrict__ x, float* __restrict__ out) {
  __shared__ __bf16 atile[16][NDIMS + 8];
  const int t    = threadIdx.x;
  const int lane = t & 31;
  const int wv   = t >> 5;
  const int hi   = lane >> 4;
  const int l15  = lane & 15;
  const int nt   = blockIdx.x * 8 + wv;             // 3125 N-tiles of 16
  const int m0   = blockIdx.y * 16;
  const bool active = (nt < NITEMS / 16);

  // cooperative stage: thread t loads 16 f32 of xA, converts, stores bf16
  {
    const int r  = t >> 4;
    const int c0 = (t & 15) * 16;
    const float* src = xA + (size_t)(m0 + r) * NDIMS + c0;
    v4f f0 = *(const v4f*)(src);
    v4f f1 = *(const v4f*)(src + 4);
    v4f f2 = *(const v4f*)(src + 8);
    v4f f3 = *(const v4f*)(src + 12);
    v8bf b0, b1;
    #pragma unroll
    for (int j = 0; j < 4; ++j) {
      b0[j]     = f2bf(f0[j]);
      b0[4 + j] = f2bf(f1[j]);
      b1[j]     = f2bf(f2[j]);
      b1[4 + j] = f2bf(f3[j]);
    }
    *(v8bf*)(&atile[r][c0])     = b0;
    *(v8bf*)(&atile[r][c0 + 8]) = b1;
  }
  __syncthreads();
  if (!active) return;                              // wave-uniform exit

  const __bf16* brow = Am + (size_t)(nt * 16 + l15) * NDIMS;

  v8f c = {};
  #pragma unroll
  for (int d = 0; d < NDIMS; d += 32) {
    const __bf16* ap = &atile[l15][d + hi * 16];
    v8bf a0 = *(const v8bf*)ap;
    v8bf a1 = *(const v8bf*)(ap + 8);
    v16bf a = cat8(a0, a1);
    v16bf b = *(const v16bf*)(brow + d + hi * 16);
    c = __builtin_amdgcn_wmma_f32_16x16x32_bf16(false, a, false, b, (short)0, c, false, false);
  }

  #pragma unroll
  for (int r = 0; r < 8; ++r) {
    const int row = m0 + r + 8 * hi;
    const size_t idx = (size_t)row * NITEMS + nt * 16 + l15;
    out[idx] = fmaxf(c[r] - x[idx], 0.0f);
  }
}

// ---------------------------------------------------------------------------
// Launch. Workspace layout (bytes):
//   [0,            25.6e6)  Am   bf16 [50000,256]
//   [25.6e6,       51.2e6)  AmT  bf16 [256,50000]
//   [51.2e6,    59.589e6)   part f32  [KSPLIT,1024,256]
//   [59.589e6,  60.637e6)   xA   f32  [1024,256]
// ---------------------------------------------------------------------------
extern "C" void kernel_launch(void* const* d_in, const int* in_sizes, int n_in,
                              void* d_out, int out_size, void* d_ws, size_t ws_size,
                              hipStream_t stream) {
  (void)in_sizes; (void)n_in; (void)out_size; (void)ws_size;
  const float* x = (const float*)d_in[0];
  const float* A = (const float*)d_in[1];
  float* out     = (float*)d_out;

  char* ws = (char*)d_ws;
  const size_t szAm = (size_t)NITEMS * NDIMS * sizeof(__bf16);       // 25,600,000
  __bf16* Am   = (__bf16*)(ws);
  __bf16* AmT  = (__bf16*)(ws + szAm);
  float*  part = (float*)(ws + 2 * szAm);
  float*  xA   = (float*)(ws + 2 * szAm + (size_t)KSPLIT * BATCH * NDIMS * sizeof(float));

  k_mask_norm<<<NITEMS / 8, 256, 0, stream>>>(A, Am);
  k_transpose<<<NITEMS / 16, 256, 0, stream>>>(Am, AmT);
  k_gemm1<<<dim3(BATCH / 16, KSPLIT), 256, 0, stream>>>(x, AmT, part);
  k_reduce<<<(BATCH * NDIMS) / 256, 256, 0, stream>>>(part, xA);
  k_gemm2<<<dim3((NITEMS / 16 + 7) / 8, BATCH / 16), 256, 0, stream>>>(xA, Am, x, out);
}
